// LinearTemporalCrossAttention_80814104641823
// MI455X (gfx1250) — matrix operand order repaired and checked by reading, not compiled, not run on hardware
//
#include <hip/hip_runtime.h>
#include <hip/hip_bf16.h>

typedef __attribute__((ext_vector_type(16))) _Float16 v16h;
typedef __attribute__((ext_vector_type(8)))  float    v8f;
typedef int v4i_ __attribute__((vector_size(16)));
typedef __attribute__((address_space(1))) v4i_* as1v4i;
typedef __attribute__((address_space(3))) v4i_* as3v4i;

#define B_   32
#define T_   1024
#define D_   1024
#define N_   77
#define NP_  96
#define L_   768
#define H_   8
#define HD_  128
#define TED_ 2048

// ---------------------------------------------------------------------------
// CDNA5 async global->LDS support (guarded; falls back to load+ds_store)
// ---------------------------------------------------------------------------
#if defined(__gfx1250__) && __has_builtin(__builtin_amdgcn_global_load_async_to_lds_b128)
#define HAS_ASYNC_LDS 1
#else
#define HAS_ASYNC_LDS 0
#endif

__device__ __forceinline__ void async_wait0() {
#if defined(__gfx1250__)
#if __has_builtin(__builtin_amdgcn_s_wait_asynccnt)
  __builtin_amdgcn_s_wait_asynccnt(0);
#else
  asm volatile("s_wait_asynccnt 0x0" ::: "memory");
#endif
#endif
}
__device__ __forceinline__ void async_wait4() {
#if defined(__gfx1250__)
#if __has_builtin(__builtin_amdgcn_s_wait_asynccnt)
  __builtin_amdgcn_s_wait_asynccnt(4);
#else
  asm volatile("s_wait_asynccnt 0x4" ::: "memory");
#endif
#endif
}

// ---------------------------------------------------------------------------
// block reduction helper (256 threads = 8 waves, wave32)
// ---------------------------------------------------------------------------
__device__ __forceinline__ float block_reduce_sum(float v, float* sbuf) {
#pragma unroll
  for (int o = 16; o > 0; o >>= 1) v += __shfl_xor(v, o, 32);
  int lane = threadIdx.x & 31, w = threadIdx.x >> 5;
  if (lane == 0) sbuf[w] = v;
  __syncthreads();
  float t = (threadIdx.x < 8) ? sbuf[threadIdx.x] : 0.0f;
  if (w == 0) {
#pragma unroll
    for (int o = 4; o > 0; o >>= 1) t += __shfl_xor(t, o, 32);
    if (lane == 0) sbuf[0] = t;
  }
  __syncthreads();
  float r = sbuf[0];
  __syncthreads();
  return r;
}

__device__ __forceinline__ float silu_f(float x) {
  return x / (1.0f + __expf(-x));
}

// ---------------------------------------------------------------------------
// Weight transpose + cast: W f32 [K][N] -> Wt f16 [N][K]
// ---------------------------------------------------------------------------
__global__ __launch_bounds__(256) void wcast_kernel(const float* __restrict__ W,
                                                    _Float16* __restrict__ Wt,
                                                    int K, int N) {
  long idx = (long)blockIdx.x * 256 + threadIdx.x;
  long total = (long)K * N;
  if (idx >= total) return;
  int k = (int)(idx / N);
  int n = (int)(idx - (long)k * N);
  Wt[(long)n * K + k] = (_Float16)W[idx];
}

// ---------------------------------------------------------------------------
// LayerNorm + cast to f16 (one block per row)
// ---------------------------------------------------------------------------
__global__ __launch_bounds__(256) void ln_cast_kernel(const float* __restrict__ X,
                                                      const float* __restrict__ g,
                                                      const float* __restrict__ bb,
                                                      _Float16* __restrict__ Y,
                                                      int D) {
  __shared__ float sbuf[8];
  long row = blockIdx.x;
  const float* xr = X + row * (long)D;
  float s = 0.f, ss = 0.f;
  for (int i = threadIdx.x; i < D; i += 256) { float v = xr[i]; s += v; ss += v * v; }
  s  = block_reduce_sum(s, sbuf);
  ss = block_reduce_sum(ss, sbuf);
  float mean = s / (float)D;
  float var  = ss / (float)D - mean * mean;
  float inv  = rsqrtf(var + 1e-5f);
  _Float16* yr = Y + row * (long)D;
  for (int i = threadIdx.x; i < D; i += 256)
    yr[i] = (_Float16)((xr[i] - mean) * inv * g[i] + bb[i]);
}

// LN(xf) with N padded 77 -> 96 (pad rows zero-filled)
__global__ __launch_bounds__(256) void ln_xf_kernel(const float* __restrict__ XF,
                                                    const float* __restrict__ g,
                                                    const float* __restrict__ bb,
                                                    _Float16* __restrict__ Y) {
  int rowOut = blockIdx.x;           // 0 .. B*NP-1
  int b = rowOut / NP_, n = rowOut - b * NP_;
  _Float16* yr = Y + (long)rowOut * L_;
  if (n >= N_) {
    for (int i = threadIdx.x; i < L_; i += 256) yr[i] = (_Float16)0.0f;
    return;
  }
  __shared__ float sbuf[8];
  const float* xr = XF + ((long)b * N_ + n) * L_;
  float s = 0.f, ss = 0.f;
  for (int i = threadIdx.x; i < L_; i += 256) { float v = xr[i]; s += v; ss += v * v; }
  s  = block_reduce_sum(s, sbuf);
  ss = block_reduce_sum(ss, sbuf);
  float mean = s / (float)L_;
  float var  = ss / (float)L_ - mean * mean;
  float inv  = rsqrtf(var + 1e-5f);
  for (int i = threadIdx.x; i < L_; i += 256)
    yr[i] = (_Float16)((xr[i] - mean) * inv * g[i] + bb[i]);
}

// ---------------------------------------------------------------------------
// Tiled WMMA GEMM:  C[M,N] = A16[M,K] * Wt16[N,K]^T (+ bias[col]) (+ resid)
// BM=BN=128, BK=32, 256 threads = 8 waves, wave tile 32x64 (2x4 wmma frags)
// grid.z batching: offsets = (z/zmod)*Out + (z%zmod)*In (element units)
// out_f16: store D as f16 instead of f32
// ---------------------------------------------------------------------------
__global__ __launch_bounds__(256) void wmma_gemm_kernel(
    const _Float16* __restrict__ A, long lda,
    const _Float16* __restrict__ Wt, long ldw,
    void* __restrict__ Cv, long ldc,
    int K,
    const float* __restrict__ bias,
    const float* __restrict__ resid,
    int out_f16,
    int zmod, long aOut, long aIn, long wOut, long wIn, long cOut, long cIn) {
  int z  = blockIdx.z;
  int zo = z / zmod, zi = z - zo * zmod;
  A  += zo * aOut + zi * aIn;
  Wt += zo * wOut + zi * wIn;
  const long cOff = zo * cOut + zi * cIn;

  const int tid  = threadIdx.x;
  const int lane = tid & 31;
  const int wave = tid >> 5;
  const int wm   = wave & 3;      // 4 waves over 128 rows
  const int wn   = wave >> 2;     // 2 waves over 128 cols
  const int rowBase = blockIdx.y * 128;
  const int colBase = blockIdx.x * 128;

  const v8f vzero = {0.f, 0.f, 0.f, 0.f, 0.f, 0.f, 0.f, 0.f};
  v8f acc[2][4];
#pragma unroll
  for (int i = 0; i < 2; ++i)
#pragma unroll
    for (int j = 0; j < 4; ++j) acc[i][j] = vzero;

  const int sr = tid >> 1;            // 0..127 : staging row/col
  const int sk = (tid & 1) * 16;      // 0 or 16 : staging k-chunk

  union U16 { uint4 q[2]; v16h v; };
  const int hl   = lane >> 4;         // 0/1 half-wave select
  const int lrow = lane & 15;

#if HAS_ASYNC_LDS
  // ---------------- double-buffered ASYNCcnt staging -----------------------
  __shared__ __align__(16) _Float16 sA[2][128 * 32];
  __shared__ __align__(16) _Float16 sB[2][128 * 32];

#define STAGE_ASYNC(bufsel, kk0)                                               \
  {                                                                            \
    const _Float16* ga_ = A + (long)(rowBase + sr) * lda + (kk0) + sk;         \
    const _Float16* gb_ = Wt + (long)(colBase + sr) * ldw + (kk0) + sk;        \
    _Float16* la_ = &sA[bufsel][sr * 32 + sk];                                 \
    _Float16* lb_ = &sB[bufsel][sr * 32 + sk];                                 \
    __builtin_amdgcn_global_load_async_to_lds_b128(                            \
        (as1v4i)(void*)ga_, (as3v4i)(void*)la_, 0, 0);                         \
    __builtin_amdgcn_global_load_async_to_lds_b128(                            \
        (as1v4i)(void*)(ga_ + 8), (as3v4i)(void*)(la_ + 8), 0, 0);             \
    __builtin_amdgcn_global_load_async_to_lds_b128(                            \
        (as1v4i)(void*)gb_, (as3v4i)(void*)lb_, 0, 0);                         \
    __builtin_amdgcn_global_load_async_to_lds_b128(                            \
        (as1v4i)(void*)(gb_ + 8), (as3v4i)(void*)(lb_ + 8), 0, 0);             \
  }

  STAGE_ASYNC(0, 0);
  int cur = 0;
  for (int k0 = 0; k0 < K; k0 += 32) {
    const bool hasNext = (k0 + 32 < K);
    if (hasNext) STAGE_ASYNC(cur ^ 1, k0 + 32);
    if (hasNext) async_wait4(); else async_wait0();
    __syncthreads();                       // current tile visible to all waves

    v16h af[2];
#pragma unroll
    for (int mi = 0; mi < 2; ++mi) {
      int m = wm * 32 + mi * 16 + lrow;
      const uint4* p = (const uint4*)(&sA[cur][m * 32]);
      U16 u; u.q[0] = p[hl]; u.q[1] = p[2 + hl];
      af[mi] = u.v;
    }
    v16h bf[4];
#pragma unroll
    for (int ni = 0; ni < 4; ++ni) {
      int n = wn * 64 + ni * 16 + lrow;
      const uint4* p = (const uint4*)(&sB[cur][n * 32]);
      U16 u; u.q[0] = p[hl]; u.q[1] = p[2 + hl];
      bf[ni] = u.v;
    }
#pragma unroll
    for (int mi = 0; mi < 2; ++mi)
#pragma unroll
      for (int ni = 0; ni < 4; ++ni)
        acc[mi][ni] = __builtin_amdgcn_wmma_f32_16x16x32_f16(
            false, af[mi], false, bf[ni], (short)0, acc[mi][ni], false, false);
    __syncthreads();                       // all reads done before restage
    cur ^= 1;
  }
#undef STAGE_ASYNC
#else
  // ---------------- fallback: sync load + ds_store staging -----------------
  __shared__ __align__(16) _Float16 sA1[128 * 32];
  __shared__ __align__(16) _Float16 sB1[128 * 32];
  for (int k0 = 0; k0 < K; k0 += 32) {
    __syncthreads();
    {
      const uint4* ga = (const uint4*)(A + (long)(rowBase + sr) * lda + k0 + sk);
      uint4* la = (uint4*)(sA1 + sr * 32 + sk);
      la[0] = ga[0]; la[1] = ga[1];
      const uint4* gb = (const uint4*)(Wt + (long)(colBase + sr) * ldw + k0 + sk);
      uint4* lb = (uint4*)(sB1 + sr * 32 + sk);
      lb[0] = gb[0]; lb[1] = gb[1];
      if (k0 + 32 < K) {
        __builtin_prefetch(A  + (long)(rowBase + sr) * lda + k0 + 32 + sk, 0, 0);
        __builtin_prefetch(Wt + (long)(colBase + sr) * ldw + k0 + 32 + sk, 0, 0);
      }
    }
    __syncthreads();
    v16h af[2];
#pragma unroll
    for (int mi = 0; mi < 2; ++mi) {
      int m = wm * 32 + mi * 16 + lrow;
      const uint4* p = (const uint4*)(sA1 + m * 32);
      U16 u; u.q[0] = p[hl]; u.q[1] = p[2 + hl];
      af[mi] = u.v;
    }
    v16h bf[4];
#pragma unroll
    for (int ni = 0; ni < 4; ++ni) {
      int n = wn * 64 + ni * 16 + lrow;
      const uint4* p = (const uint4*)(sB1 + n * 32);
      U16 u; u.q[0] = p[hl]; u.q[1] = p[2 + hl];
      bf[ni] = u.v;
    }
#pragma unroll
    for (int mi = 0; mi < 2; ++mi)
#pragma unroll
      for (int ni = 0; ni < 4; ++ni)
        acc[mi][ni] = __builtin_amdgcn_wmma_f32_16x16x32_f16(
            false, af[mi], false, bf[ni], (short)0, acc[mi][ni], false, false);
  }
#endif

  // epilogue: C/D layout — lane l holds col l%16, rows r + 8*(l>=16)
  const int colLane = lane & 15;
  const int rowLane = hl * 8;
  float* Cf = (float*)Cv + cOff;
  _Float16* Ch = (_Float16*)Cv + cOff;
#pragma unroll
  for (int mi = 0; mi < 2; ++mi)
#pragma unroll
    for (int ni = 0; ni < 4; ++ni) {
      int col = colBase + wn * 64 + ni * 16 + colLane;
      float bval = bias ? bias[col] : 0.0f;
#pragma unroll
      for (int r = 0; r < 8; ++r) {
        int row = rowBase + wm * 32 + mi * 16 + rowLane + r;
        long ci = (long)row * ldc + col;
        float vv = acc[mi][ni][r] + bval;
        if (resid) vv += resid[ci];
        if (out_f16) Ch[ci] = (_Float16)vv;
        else         Cf[ci] = vv;
      }
    }
}

// ---------------------------------------------------------------------------
// softmax over head_dim (128) per (row, head); 8 waves = 8 heads per row
// ---------------------------------------------------------------------------
__global__ __launch_bounds__(256) void qsoftmax_kernel(const float* __restrict__ Q,
                                                       _Float16* __restrict__ Q16) {
  long row = blockIdx.x;
  int wave = threadIdx.x >> 5, lane = threadIdx.x & 31;
  const float* qr = Q + row * (long)D_ + wave * HD_;
  float v[4];
  float m = -1e30f;
#pragma unroll
  for (int j = 0; j < 4; ++j) { v[j] = qr[lane + j * 32]; m = fmaxf(m, v[j]); }
#pragma unroll
  for (int o = 16; o > 0; o >>= 1) m = fmaxf(m, __shfl_xor(m, o, 32));
  float s = 0.f;
#pragma unroll
  for (int j = 0; j < 4; ++j) { v[j] = __expf(v[j] - m); s += v[j]; }
#pragma unroll
  for (int o = 16; o > 0; o >>= 1) s += __shfl_xor(s, o, 32);
  float inv = 1.0f / s;
  _Float16* yr = Q16 + row * (long)D_ + wave * HD_;
#pragma unroll
  for (int j = 0; j < 4; ++j) yr[lane + j * 32] = (_Float16)(v[j] * inv);
}

// softmax over key tokens N=77 per (b,col) -> transposed f16 [b,h,d,NP]
__global__ __launch_bounds__(256) void ksoftmaxT_kernel(const float* __restrict__ Kf,
                                                        _Float16* __restrict__ KT) {
  int gid = blockIdx.x * 256 + threadIdx.x;      // B*D = 32768
  int b = gid >> 10, col = gid & 1023;
  int h = col >> 7, d = col & 127;
  const float* kp = Kf + (long)b * NP_ * D_ + col;
  float m = -1e30f;
  for (int n = 0; n < N_; ++n) m = fmaxf(m, kp[(long)n * D_]);
  float s = 0.f;
  for (int n = 0; n < N_; ++n) s += __expf(kp[(long)n * D_] - m);
  float inv = 1.0f / s;
  _Float16* op = KT + ((long)(b * H_ + h) * HD_ + d) * NP_;
  for (int n = 0; n < N_; ++n) op[n] = (_Float16)(__expf(kp[(long)n * D_] - m) * inv);
  for (int n = N_; n < NP_; ++n) op[n] = (_Float16)0.0f;
}

// v f32 [b,n,col] -> transposed f16 [b,h,l,NP], pads zeroed
__global__ __launch_bounds__(256) void castvT_kernel(const float* __restrict__ Vf,
                                                     _Float16* __restrict__ VT) {
  int gid = blockIdx.x * 256 + threadIdx.x;      // B*D = 32768
  int b = gid >> 10, col = gid & 1023;
  int h = col >> 7, l = col & 127;
  const float* vp = Vf + (long)b * NP_ * D_ + col;
  _Float16* op = VT + ((long)(b * H_ + h) * HD_ + l) * NP_;
  for (int n = 0; n < N_; ++n) op[n] = (_Float16)vp[(long)n * D_];
  for (int n = N_; n < NP_; ++n) op[n] = (_Float16)0.0f;
}

// ---------------------------------------------------------------------------
// emb_out[b,j] = silu(emb[b,:]) @ emb_W[:,j] + emb_b[j]
// ---------------------------------------------------------------------------
__global__ __launch_bounds__(256) void emb_kernel(const float* __restrict__ emb,
                                                  const float* __restrict__ W,
                                                  const float* __restrict__ bias,
                                                  float* __restrict__ outp) {
  int gid = blockIdx.x * 256 + threadIdx.x;     // B * 2D = 65536
  int b = gid >> 11, j = gid & 2047;
  const float* er = emb + (long)b * TED_;
  float acc = bias[j];
  for (int i = 0; i < TED_; ++i)
    acc += silu_f(er[i]) * W[(long)i * (2 * D_) + j];
  outp[gid] = acc;
}

// ---------------------------------------------------------------------------
// stylization: Hin = silu( LN(y)*(1+scale) + shift ) -> f16
// ---------------------------------------------------------------------------
__global__ __launch_bounds__(256) void stylize_kernel(const float* __restrict__ Y,
                                                      const float* __restrict__ eo,
                                                      const float* __restrict__ g,
                                                      const float* __restrict__ bb,
                                                      _Float16* __restrict__ Hout) {
  __shared__ float sbuf[8];
  long row = blockIdx.x;
  int b = (int)(row >> 10);                     // row / T
  const float* yr = Y + row * (long)D_;
  float s = 0.f, ss = 0.f;
  for (int i = threadIdx.x; i < D_; i += 256) { float v = yr[i]; s += v; ss += v * v; }
  s  = block_reduce_sum(s, sbuf);
  ss = block_reduce_sum(ss, sbuf);
  float mean = s / (float)D_;
  float var  = ss / (float)D_ - mean * mean;
  float inv  = rsqrtf(var + 1e-5f);
  const float* sc = eo + (long)b * (2 * D_);
  const float* sh = sc + D_;
  _Float16* hr = Hout + row * (long)D_;
  for (int i = threadIdx.x; i < D_; i += 256) {
    float hn = (yr[i] - mean) * inv * g[i] + bb[i];
    float v  = hn * (1.0f + sc[i]) + sh[i];
    hr[i] = (_Float16)silu_f(v);
  }
}

// ---------------------------------------------------------------------------
extern "C" void kernel_launch(void* const* d_in, const int* in_sizes, int n_in,
                              void* d_out, int out_size, void* d_ws, size_t ws_size,
                              hipStream_t stream) {
  const float* x       = (const float*)d_in[0];
  const float* xf      = (const float*)d_in[1];
  const float* emb     = (const float*)d_in[2];
  const float* norm_g  = (const float*)d_in[3];
  const float* norm_b  = (const float*)d_in[4];
  const float* tnorm_g = (const float*)d_in[5];
  const float* tnorm_b = (const float*)d_in[6];
  const float* Wq      = (const float*)d_in[7];
  const float* bq      = (const float*)d_in[8];
  const float* Wk      = (const float*)d_in[9];
  const float* bk      = (const float*)d_in[10];
  const float* Wv      = (const float*)d_in[11];
  const float* bv      = (const float*)d_in[12];
  const float* emb_W   = (const float*)d_in[13];
  const float* emb_b   = (const float*)d_in[14];
  const float* onorm_g = (const float*)d_in[15];
  const float* onorm_b = (const float*)d_in[16];
  const float* Wo      = (const float*)d_in[17];
  const float* bo      = (const float*)d_in[18];
  float* outp = (float*)d_out;

  char* ws = (char*)d_ws;
  size_t off = 0;
  auto alloc = [&](size_t bytes) -> void* {
    void* p = ws + off;
    off = (off + bytes + 255) & ~(size_t)255;
    return p;
  };

  const long MR = (long)B_ * T_;                        // 32768 rows
  _Float16* aq   = (_Float16*)alloc(MR * D_ * 2);       // LN(x) f16; reused as Hin16
  float*    qf   = (float*)   alloc(MR * D_ * 4);       // q f32; reused as y f32
  _Float16* q16  = (_Float16*)alloc(MR * D_ * 2);       // softmax(q) f16
  _Float16* xfn  = (_Float16*)alloc((long)B_ * NP_ * L_ * 2);
  float*    kf   = (float*)   alloc((long)B_ * NP_ * D_ * 4);
  float*    vf   = (float*)   alloc((long)B_ * NP_ * D_ * 4);
  _Float16* kT   = (_Float16*)alloc((long)B_ * H_ * HD_ * NP_ * 2); // [b,h,d,n]
  _Float16* vT   = (_Float16*)alloc((long)B_ * H_ * HD_ * NP_ * 2); // [b,h,l,n]
  _Float16* at16 = (_Float16*)alloc((long)B_ * H_ * HD_ * HD_ * 2); // [b,h,l,d]
  _Float16* wqT  = (_Float16*)alloc((long)D_ * D_ * 2);
  _Float16* wkT  = (_Float16*)alloc((long)L_ * D_ * 2);
  _Float16* wvT  = (_Float16*)alloc((long)L_ * D_ * 2);
  _Float16* woT  = (_Float16*)alloc((long)D_ * D_ * 2);
  float*    eo   = (float*)   alloc((long)B_ * 2 * D_ * 4);

  // 1) weight transpose-casts
  wcast_kernel<<<(D_ * D_ + 255) / 256, 256, 0, stream>>>(Wq, wqT, D_, D_);
  wcast_kernel<<<(L_ * D_ + 255) / 256, 256, 0, stream>>>(Wk, wkT, L_, D_);
  wcast_kernel<<<(L_ * D_ + 255) / 256, 256, 0, stream>>>(Wv, wvT, L_, D_);
  wcast_kernel<<<(D_ * D_ + 255) / 256, 256, 0, stream>>>(Wo, woT, D_, D_);

  // 2) LN(x) -> f16
  ln_cast_kernel<<<(int)MR, 256, 0, stream>>>(x, norm_g, norm_b, aq, D_);

  // 3) q = LN(x) @ Wq + bq
  wmma_gemm_kernel<<<dim3(D_ / 128, (int)(MR / 128), 1), 256, 0, stream>>>(
      aq, D_, wqT, D_, (void*)qf, D_, D_, bq, nullptr, 0,
      1, 0, 0, 0, 0, 0, 0);

  // 4) softmax over head_dim -> q16
  qsoftmax_kernel<<<(int)MR, 256, 0, stream>>>(qf, q16);

  // 5) LN(xf) padded -> xfn16
  ln_xf_kernel<<<B_ * NP_, 256, 0, stream>>>(xf, tnorm_g, tnorm_b, xfn);

  // 6/7) k, v projections (M = B*NP = 3072)
  wmma_gemm_kernel<<<dim3(D_ / 128, (B_ * NP_) / 128, 1), 256, 0, stream>>>(
      xfn, L_, wkT, L_, (void*)kf, D_, L_, bk, nullptr, 0,
      1, 0, 0, 0, 0, 0, 0);
  wmma_gemm_kernel<<<dim3(D_ / 128, (B_ * NP_) / 128, 1), 256, 0, stream>>>(
      xfn, L_, wvT, L_, (void*)vf, D_, L_, bv, nullptr, 0,
      1, 0, 0, 0, 0, 0, 0);

  // 8) key-token softmax -> kT (f16, [b,h,d,n], pads 0); v -> vT
  ksoftmaxT_kernel<<<(B_ * D_) / 256, 256, 0, stream>>>(kf, kT);
  castvT_kernel<<<(B_ * D_) / 256, 256, 0, stream>>>(vf, vT);

  // 9) attnT[b,h,l,d] = sum_n v[n,l]*k[n,d] via WMMA (M=N=128, K=NP=96)
  wmma_gemm_kernel<<<dim3(1, 1, B_ * H_), 256, 0, stream>>>(
      vT, NP_, kT, NP_, (void*)at16, HD_, NP_, nullptr, nullptr, 1,
      1, (long)HD_ * NP_, 0, (long)HD_ * NP_, 0, (long)HD_ * HD_, 0);

  // 10) y = q16 @ attn  (batched per (b,h); y reuses qf)
  wmma_gemm_kernel<<<dim3(1, T_ / 128, B_ * H_), 256, 0, stream>>>(
      q16, D_, at16, HD_, (void*)qf, D_, HD_, nullptr, nullptr, 0,
      H_, (long)T_ * D_, HD_, (long)H_ * HD_ * HD_, (long)HD_ * HD_,
      (long)T_ * D_, HD_);

  // 11) emb_out = silu(emb) @ emb_W + emb_b
  emb_kernel<<<(B_ * 2 * D_) / 256, 256, 0, stream>>>(emb, emb_W, emb_b, eo);

  // 12) Hin = silu(LN(y)*(1+scale)+shift) -> f16 (reuses aq)
  stylize_kernel<<<(int)MR, 256, 0, stream>>>(qf, eo, onorm_g, onorm_b, aq);

  // 13) out = x + Hin @ Wo + bo
  wmma_gemm_kernel<<<dim3(D_ / 128, (int)(MR / 128), 1), 256, 0, stream>>>(
      aq, D_, woT, D_, (void*)outp, D_, D_, bo, x, 0,
      1, 0, 0, 0, 0, 0, 0);
}
